// LongNetViT_46394236732031
// MI455X (gfx1250) — matrix operand order repaired and checked by reading, not compiled, not run on hardware
//
#include <hip/hip_runtime.h>

#ifndef __has_builtin
#define __has_builtin(x) 0
#endif
#if __has_builtin(__builtin_amdgcn_tensor_load_to_lds)
#define USE_TDM 1
#else
#define USE_TDM 0
#endif

typedef __bf16 bf16;
typedef __attribute__((ext_vector_type(16))) __bf16 v16bf;
typedef __attribute__((ext_vector_type(8)))  float  v8f;
typedef __attribute__((ext_vector_type(4)))  unsigned int v4u;
typedef __attribute__((ext_vector_type(4)))  int v4i;
typedef __attribute__((ext_vector_type(8)))  int v8i;

union FragU { v16bf v; float4 f[2]; };

__device__ __forceinline__ v8f wmma_bf16(v16bf a, v16bf b, v8f c) {
  return __builtin_amdgcn_wmma_f32_16x16x32_bf16(false, a, false, b, (short)0, c, false, false);
}

// A-operand fragment: 16x32 (MxK), source row-major, ld in elements.
// Layout (ISA 7.12.2): lane = M row (lane&15); half=lane>>4 selects K chunks
// [half*8 .. half*8+7] and [16+half*8 .. 16+half*8+7].
__device__ __forceinline__ v16bf load_a(const bf16* p, int ld) {
  const int lane = threadIdx.x & 31;
  const int row = lane & 15, half = lane >> 4;
  const bf16* b = p + (size_t)row * ld + half * 8;
  FragU u;
  u.f[0] = *(const float4*)(b);
  u.f[1] = *(const float4*)(b + 16);
  return u.v;
}

__device__ __forceinline__ v16bf load_a_rc(const bf16* p, int ld, int rowcap) {
  const int lane = threadIdx.x & 31;
  int row = lane & 15;
  if (row > rowcap) row = (rowcap > 0) ? rowcap : 0;
  const int half = lane >> 4;
  const bf16* b = p + (size_t)row * ld + half * 8;
  FragU u;
  u.f[0] = *(const float4*)(b);
  u.f[1] = *(const float4*)(b + 16);
  return u.v;
}

// B-operand fragment: 32x16 (KxN), source given as row-major [N][K].
// Layout: lane = N col (lane&15); half selects K chunks half*8..+7 and 16+half*8..+7.
__device__ __forceinline__ v16bf load_b(const bf16* p, int ld) {
  const int lane = threadIdx.x & 31;
  const int col = lane & 15, half = lane >> 4;
  const bf16* b = p + (size_t)col * ld + half * 8;
  FragU u;
  u.f[0] = *(const float4*)(b);
  u.f[1] = *(const float4*)(b + 16);
  return u.v;
}

__device__ __forceinline__ void wait_tensorcnt0() {
#if __has_builtin(__builtin_amdgcn_s_wait_tensorcnt)
  __builtin_amdgcn_s_wait_tensorcnt(0);
#else
  asm volatile("s_wait_tensorcnt 0x0" ::: "memory");
#endif
}

#if USE_TDM
// Issue one TDM descriptor: load a [rows-capped 128] x 32 bf16 tile (row-major,
// leading dim K elements) from gptr into LDS at lds_off, padding each 64B row
// with 32B (LDS leading dim = 48 elements) per D# pad fields.
__device__ __forceinline__ void tdm_load_tile(unsigned lds_off, const void* gptr,
                                              int K, int rows) {
  const unsigned long long ga = (unsigned long long)gptr;
  v4u g0;
  g0.x = 1u;                                    // count=1, user-mode descriptor
  g0.y = lds_off;                               // lds_addr (bytes)
  g0.z = (unsigned)(ga & 0xFFFFFFFFu);          // global_addr[31:0]
  g0.w = (unsigned)((ga >> 32) & 0x1FFFFFFu) | (2u << 30);   // addr[56:32], type=2
  v8i g1;
  // data_size=1 (2B), pad_enable, pad_interval=3 (16 DW = 64B), pad_amount=7 (8 DW = 32B)
  g1[0] = (1 << 16) | (1 << 20) | (3 << 22) | (7 << 25);
  g1[1] = (K & 0xFFFF) << 16;                   // tensor_dim0[15:0] @ bits 63:48
  g1[2] = ((unsigned)K >> 16) | ((rows & 0xFFFF) << 16); // dim0 hi, tensor_dim1 lo
  g1[3] = ((unsigned)rows >> 16) | (32 << 16);  // dim1 hi, tile_dim0=32
  g1[4] = 128;                                  // tile_dim1=128, tile_dim2=0
  g1[5] = K;                                    // tensor_dim0_stride[31:0]
  g1[6] = 0;
  g1[7] = 0;
  const v4i z4 = {0, 0, 0, 0};
#if __clang_major__ >= 23
  const v8i z8 = {0, 0, 0, 0, 0, 0, 0, 0};
  __builtin_amdgcn_tensor_load_to_lds(g0, g1, z4, z4, z8, 0);
#else
  __builtin_amdgcn_tensor_load_to_lds(g0, g1, z4, z4, 0);
#endif
}

__device__ __forceinline__ unsigned lds_addr_of(const void* p) {
  return (unsigned)(unsigned long long)p;   // low 32 bits of LDS-aperture address
}
#endif

// ---------------------------------------------------------------------------
// GEMM: out[M,N] = act(A[M,K] @ W[N,K]^T + bias) (*gamma) (+res), out f32/bf16
// TDM double-buffered tile pipeline (fallback: cooperative global->LDS copy).
// ---------------------------------------------------------------------------
__global__ void __launch_bounds__(256) gemm_wmma_k(
    const bf16* __restrict__ A, const bf16* __restrict__ W,
    const float* __restrict__ bias, const float* __restrict__ gam,
    const float* __restrict__ res, float* __restrict__ outF, bf16* __restrict__ outB,
    int M, int N, int K, int act_gelu)
{
#if USE_TDM
  __shared__ bf16 As[2][128 * 48];
  __shared__ bf16 Ws[2][128 * 48];
#else
  __shared__ bf16 As[1][128 * 48];
  __shared__ bf16 Ws[1][128 * 48];
#endif
  const int tid = threadIdx.x, lane = tid & 31, wid = tid >> 5;
  const int wm = wid >> 2, wn = wid & 3;
  const int m0 = blockIdx.y * 128, n0 = blockIdx.x * 128;

  v8f acc[4][2];
#pragma unroll
  for (int i = 0; i < 4; ++i)
#pragma unroll
    for (int j = 0; j < 2; ++j)
#pragma unroll
      for (int e = 0; e < 8; ++e) acc[i][j][e] = 0.f;

#if USE_TDM
  const bf16* abase = A + (size_t)m0 * K;
  const bf16* wbase = W + (size_t)n0 * K;
  const int arows = M - m0, wrows = N - n0;
  if (wid == 0) {
    tdm_load_tile(lds_addr_of(&As[0][0]), abase, K, arows);
    tdm_load_tile(lds_addr_of(&Ws[0][0]), wbase, K, wrows);
    wait_tensorcnt0();
  }
  __syncthreads();
  int buf = 0;
  for (int k0 = 0; k0 < K; k0 += 32) {
    if (wid == 0 && k0 + 32 < K) {
      tdm_load_tile(lds_addr_of(&As[buf ^ 1][0]), abase + k0 + 32, K, arows);
      tdm_load_tile(lds_addr_of(&Ws[buf ^ 1][0]), wbase + k0 + 32, K, wrows);
    }
    v16bf af[4], wf[2];
#pragma unroll
    for (int i = 0; i < 4; ++i) af[i] = load_a(&As[buf][(wm * 64 + i * 16) * 48], 48);
#pragma unroll
    for (int j = 0; j < 2; ++j) wf[j] = load_b(&Ws[buf][(wn * 32 + j * 16) * 48], 48);
#pragma unroll
    for (int i = 0; i < 4; ++i)
#pragma unroll
      for (int j = 0; j < 2; ++j)
        acc[i][j] = wmma_bf16(af[i], wf[j], acc[i][j]);
    if (wid == 0) wait_tensorcnt0();
    __syncthreads();
    buf ^= 1;
  }
#else
  const int lr = tid >> 1;
  const int lc = (tid & 1) * 16;
  int arow = m0 + lr; if (arow >= M) arow = M - 1;
  int wrow = n0 + lr; if (wrow >= N) wrow = N - 1;
  const bf16* ap = A + (size_t)arow * K;
  const bf16* wp = W + (size_t)wrow * K;
  for (int k0 = 0; k0 < K; k0 += 32) {
    __syncthreads();
    *(float4*)&As[0][lr * 48 + lc]     = *(const float4*)(ap + k0 + lc);
    *(float4*)&As[0][lr * 48 + lc + 8] = *(const float4*)(ap + k0 + lc + 8);
    *(float4*)&Ws[0][lr * 48 + lc]     = *(const float4*)(wp + k0 + lc);
    *(float4*)&Ws[0][lr * 48 + lc + 8] = *(const float4*)(wp + k0 + lc + 8);
    if (k0 + 32 < K) {
      __builtin_prefetch(ap + k0 + 32 + lc, 0, 1);
      __builtin_prefetch(wp + k0 + 32 + lc, 0, 1);
    }
    __syncthreads();
    v16bf af[4], wf[2];
#pragma unroll
    for (int i = 0; i < 4; ++i) af[i] = load_a(&As[0][(wm * 64 + i * 16) * 48], 48);
#pragma unroll
    for (int j = 0; j < 2; ++j) wf[j] = load_b(&Ws[0][(wn * 32 + j * 16) * 48], 48);
#pragma unroll
    for (int i = 0; i < 4; ++i)
#pragma unroll
      for (int j = 0; j < 2; ++j)
        acc[i][j] = wmma_bf16(af[i], wf[j], acc[i][j]);
  }
#endif

  const int half = lane >> 4, cn = lane & 15;
#pragma unroll
  for (int i = 0; i < 4; ++i) {
#pragma unroll
    for (int j = 0; j < 2; ++j) {
      const int nc = n0 + wn * 32 + j * 16 + cn;
#pragma unroll
      for (int e = 0; e < 8; ++e) {
        const int m = m0 + wm * 64 + i * 16 + half * 8 + e;
        if (m < M && nc < N) {
          float v = acc[i][j][e] + bias[nc];
          if (act_gelu) v = 0.5f * v * (1.f + erff(v * 0.70710678118f));
          if (gam) v *= gam[nc];
          if (res) v += res[(size_t)m * N + nc];
          if (outF) outF[(size_t)m * N + nc] = v;
          if (outB) outB[(size_t)m * N + nc] = (bf16)v;
        }
      }
    }
  }
}

// ---------------------------------------------------------------------------
// Flash attention (H=8, DH=64) with optional dilated-token indexing.
// Token rows laid out [B, T, H, DH]. For dilated: token = s*w + h%r + r*j.
// ---------------------------------------------------------------------------
__global__ void __launch_bounds__(128) flash_attn_k(
    const bf16* __restrict__ Qp, const bf16* __restrict__ Kp, const bf16* __restrict__ Vp,
    bf16* __restrict__ Op, float* __restrict__ Lse,
    int Tq, int Tkv, int qlen, int kvlen, int wseg, int rdil, int dilated, float scale)
{
  __shared__ bf16 Pl[4][16 * 40];
  __shared__ bf16 Vt[64 * 40];
  const int tid = threadIdx.x, lane = tid & 31, wid = tid >> 5;
  const int b = blockIdx.z;
  int h, off, str;
  if (dilated) {
    const int s = blockIdx.y >> 3;
    h = blockIdx.y & 7;
    off = s * wseg + (h % rdil);
    str = rdil;
  } else { h = blockIdx.y; off = 0; str = 1; }

  const size_t qbase = ((size_t)((size_t)b * Tq + off) * 8 + h) * 64;
  const size_t kbase = ((size_t)((size_t)b * Tkv + off) * 8 + h) * 64;
  const int ldq = str * 512;               // H*DH per token step * dilation
  const int qi0 = blockIdx.x * 64 + wid * 16;
  const int rowcap = qlen - 1 - qi0;

  const v16bf aq0 = load_a_rc(Qp + qbase + (size_t)qi0 * ldq, ldq, rowcap);
  const v16bf aq1 = load_a_rc(Qp + qbase + (size_t)qi0 * ldq + 32, ldq, rowcap);

  v8f o[4];
#pragma unroll
  for (int t = 0; t < 4; ++t)
#pragma unroll
    for (int e = 0; e < 8; ++e) o[t][e] = 0.f;
  float m8[8], l8[8];
#pragma unroll
  for (int e = 0; e < 8; ++e) { m8[e] = -3.0e38f; l8[e] = 0.f; }

  const int half = lane >> 4, cn = lane & 15;

  for (int kb = 0; kb < kvlen; kb += 32) {
    __syncthreads();
    { // cooperative V block transpose: [32 keys x 64 dims] -> Vt[dim][key]
      const int kk = tid >> 2;
      const int d0 = (tid & 3) * 16;
      const bf16* vr = Vp + kbase + (size_t)(kb + kk) * ldq + d0;
#pragma unroll
      for (int d = 0; d < 16; ++d) Vt[(d0 + d) * 40 + kk] = vr[d];
    }
    __syncthreads();

    // S tiles: keys [kb..kb+15], [kb+16..kb+31]; chain two wmma over DH=64
    const bf16* kp0 = Kp + kbase + (size_t)kb * ldq;
    const bf16* kp1 = kp0 + (size_t)16 * ldq;
    v8f s0, s1;
#pragma unroll
    for (int e = 0; e < 8; ++e) { s0[e] = 0.f; s1[e] = 0.f; }
    s0 = wmma_bf16(aq0, load_b(kp0, ldq), s0);
    s0 = wmma_bf16(aq1, load_b(kp0 + 32, ldq), s0);
    s1 = wmma_bf16(aq0, load_b(kp1, ldq), s1);
    s1 = wmma_bf16(aq1, load_b(kp1 + 32, ldq), s1);

    // streaming softmax; rows live in 16-lane halves
#pragma unroll
    for (int e = 0; e < 8; ++e) {
      const float a0 = s0[e] * scale, a1 = s1[e] * scale;
      float rm = fmaxf(a0, a1);
      rm = fmaxf(rm, __shfl_xor(rm, 8, 16));
      rm = fmaxf(rm, __shfl_xor(rm, 4, 16));
      rm = fmaxf(rm, __shfl_xor(rm, 2, 16));
      rm = fmaxf(rm, __shfl_xor(rm, 1, 16));
      const float mn = fmaxf(m8[e], rm);
      const float al = __expf(m8[e] - mn);
      const float p0 = __expf(a0 - mn);
      const float p1 = __expf(a1 - mn);
      float rs = p0 + p1;
      rs += __shfl_xor(rs, 8, 16);
      rs += __shfl_xor(rs, 4, 16);
      rs += __shfl_xor(rs, 2, 16);
      rs += __shfl_xor(rs, 1, 16);
      l8[e] = l8[e] * al + rs;
      m8[e] = mn;
#pragma unroll
      for (int t = 0; t < 4; ++t) o[t][e] *= al;
      Pl[wid][(e + half * 8) * 40 + cn]      = (bf16)p0;
      Pl[wid][(e + half * 8) * 40 + 16 + cn] = (bf16)p1;
    }
    asm volatile("s_wait_dscnt 0x0" ::: "memory");
    const v16bf pa = load_a(&Pl[wid][0], 40);
#pragma unroll
    for (int t = 0; t < 4; ++t)
      o[t] = wmma_bf16(pa, load_b(&Vt[t * 16 * 40], 40), o[t]);
  }

#pragma unroll
  for (int e = 0; e < 8; ++e) {
    const int j = qi0 + e + half * 8;
    if (j < qlen) {
      const float inv = 1.f / l8[e];
      const size_t tok = (size_t)b * Tq + off + (size_t)str * j;
      bf16* orow = Op + (tok * 8 + h) * 64;
#pragma unroll
      for (int t = 0; t < 4; ++t) orow[t * 16 + cn] = (bf16)(o[t][e] * inv);
      if (Lse && cn == 0) Lse[tok * 8 + h] = m8[e] + __logf(l8[e]);
    }
  }
}

// ---------------------------------------------------------------------------
// LayerNorm over D=512 (one block/row), writes f32 and/or bf16
// ---------------------------------------------------------------------------
__global__ void __launch_bounds__(256) layernorm_k(
    const float* __restrict__ X, const float* __restrict__ G, const float* __restrict__ Bt,
    float* __restrict__ outF, bf16* __restrict__ outB)
{
  __shared__ float red[16];
  const int row = blockIdx.x, tid = threadIdx.x;
  const float* x = X + (size_t)row * 512;
  const float v0 = x[tid], v1 = x[tid + 256];
  float s = v0 + v1, q = v0 * v0 + v1 * v1;
#pragma unroll
  for (int o = 16; o > 0; o >>= 1) { s += __shfl_xor(s, o, 32); q += __shfl_xor(q, o, 32); }
  if ((tid & 31) == 0) { red[tid >> 5] = s; red[8 + (tid >> 5)] = q; }
  __syncthreads();
  s = 0.f; q = 0.f;
#pragma unroll
  for (int i = 0; i < 8; ++i) { s += red[i]; q += red[8 + i]; }
  const float mean = s * (1.f / 512.f);
  const float var = q * (1.f / 512.f) - mean * mean;
  const float rstd = rsqrtf(var + 1e-5f);
  const float y0 = (v0 - mean) * rstd * G[tid] + Bt[tid];
  const float y1 = (v1 - mean) * rstd * G[tid + 256] + Bt[tid + 256];
  if (outF) { outF[(size_t)row * 512 + tid] = y0; outF[(size_t)row * 512 + tid + 256] = y1; }
  if (outB) { outB[(size_t)row * 512 + tid] = (bf16)y0; outB[(size_t)row * 512 + tid + 256] = (bf16)y1; }
}

// pos-embed add: x = contexts + pos_embed[grid(coords)]
__global__ void __launch_bounds__(128) posadd_k(
    const float* __restrict__ ctx, const int* __restrict__ coords,
    const float* __restrict__ pe, float* __restrict__ xout)
{
  const int row = blockIdx.x, tid = threadIdx.x;
  const int c0 = coords[(size_t)row * 2], c1 = coords[(size_t)row * 2 + 1];
  const int pidx = (c0 >> 8) * 256 + (c1 >> 8);   // PATCH=256, NGRIDS=256
  const float* a = ctx + (size_t)row * 512;
  const float* p = pe + (size_t)pidx * 512;
  for (int i = tid; i < 512; i += 128) xout[(size_t)row * 512 + i] = a[i] + p[i];
}

// combine 5 dilated branches via softmax(lse) weights
__global__ void __launch_bounds__(256) combine_k(
    const bf16* __restrict__ bo, const float* __restrict__ bl, bf16* __restrict__ out)
{
  const size_t BLHc = (size_t)2 * 8192 * 8;
  const size_t BLDc = (size_t)2 * 8192 * 512;
  const size_t idx = (size_t)blockIdx.x * 4 + (threadIdx.x >> 6);
  const int d = threadIdx.x & 63;
  float ls[5], mx = -3.0e38f;
#pragma unroll
  for (int k = 0; k < 5; ++k) { ls[k] = bl[k * BLHc + idx]; mx = fmaxf(mx, ls[k]); }
  float sum = 0.f, acc = 0.f;
#pragma unroll
  for (int k = 0; k < 5; ++k) {
    const float wk = __expf(ls[k] - mx);
    sum += wk;
    acc += wk * (float)bo[k * BLDc + idx * 64 + d];
  }
  out[idx * 64 + d] = (bf16)(acc / sum);
}

__global__ void __launch_bounds__(128) combbuild_k(const float* __restrict__ qf,
    const float* __restrict__ inst, float* __restrict__ comb)
{
  const int row = blockIdx.x;   // B*128
  const int b = row >> 7, t = row & 127;
  const float* src = (t < 32) ? (qf + ((size_t)b * 32 + t) * 512)
                              : (inst + ((size_t)b * 96 + (t - 32)) * 512);
  for (int i = threadIdx.x; i < 512; i += 128)
    comb[(size_t)row * 512 + i] = src[i];
}

__global__ void __launch_bounds__(128) extractq_k(const float* __restrict__ comb,
    float* __restrict__ qf, bf16* __restrict__ qfb)
{
  const int row = blockIdx.x;   // B*32
  const int b = row >> 5, t = row & 31;
  const float* src = comb + ((size_t)b * 128 + t) * 512;
  for (int i = threadIdx.x; i < 512; i += 128) {
    const float v = src[i];
    qf[(size_t)row * 512 + i] = v;
    qfb[(size_t)row * 512 + i] = (bf16)v;
  }
}

__global__ void f2b_k(const float* __restrict__ a, bf16* __restrict__ o, size_t n) {
  size_t i = (size_t)blockIdx.x * blockDim.x + threadIdx.x;
  const size_t st = (size_t)gridDim.x * blockDim.x;
  for (; i < n; i += st) o[i] = (bf16)a[i];
}
__global__ void fillf_k(float* __restrict__ p, float v, size_t n) {
  size_t i = (size_t)blockIdx.x * blockDim.x + threadIdx.x;
  const size_t st = (size_t)gridDim.x * blockDim.x;
  for (; i < n; i += st) p[i] = v;
}
__global__ void copyf_k(const float* __restrict__ a, float* __restrict__ o, size_t n) {
  size_t i = (size_t)blockIdx.x * blockDim.x + threadIdx.x;
  const size_t st = (size_t)gridDim.x * blockDim.x;
  for (; i < n; i += st) o[i] = a[i];
}

// ---------------------------------------------------------------------------
extern "C" void kernel_launch(void* const* d_in, const int* in_sizes, int n_in,
                              void* d_out, int out_size, void* d_ws, size_t ws_size,
                              hipStream_t stream)
{
  (void)in_sizes; (void)n_in; (void)out_size; (void)ws_size;
  const float* in_q     = (const float*)d_in[0];
  const float* in_ctx   = (const float*)d_in[1];
  const float* in_inst  = (const float*)d_in[2];
  const int*   in_crd   = (const int*)d_in[3];
  const float* in_pe    = (const float*)d_in[4];
  const float* w_lnqkv  = (const float*)d_in[5];
  const float* b_lnqkv  = (const float*)d_in[6];
  const float* lnln     = (const float*)d_in[7];
  const float* w_fc1    = (const float*)d_in[8];
  const float* b_fc1    = (const float*)d_in[9];
  const float* w_fc2    = (const float*)d_in[10];
  const float* b_fc2    = (const float*)d_in[11];
  const float* lnfinal  = (const float*)d_in[12];
  const float* w_saqkv  = (const float*)d_in[13];
  const float* b_saqkv  = (const float*)d_in[14];
  const float* saln     = (const float*)d_in[15];
  const float* sagamma  = (const float*)d_in[16];
  const float* w_safc1  = (const float*)d_in[17];
  const float* b_safc1  = (const float*)d_in[18];
  const float* w_safc2  = (const float*)d_in[19];
  const float* b_safc2  = (const float*)d_in[20];
  const float* w_caqkv  = (const float*)d_in[21];
  const float* b_caqkv  = (const float*)d_in[22];
  const float* caln     = (const float*)d_in[23];

  const size_t BLD = 8388608;   // B*L*D
  const size_t BLH = 131072;    // B*L*H
  const size_t BLF = 33554432;  // B*L*FF

  char* wsp = (char*)d_ws; size_t woff = 0;
  auto alloc = [&](size_t bytes) -> void* {
    void* p = wsp + woff; woff = (woff + bytes + 255) & ~(size_t)255; return p;
  };
  float* xbuf = (float*)alloc(BLD * 4);
  bf16* hb   = (bf16*)alloc(BLD * 2);
  bf16* qb   = (bf16*)alloc(BLD * 2);
  bf16* kb   = (bf16*)alloc(BLD * 2);
  bf16* vb   = (bf16*)alloc(BLD * 2);
  bf16* bro  = (bf16*)alloc(5 * BLD * 2);
  float* brl = (float*)alloc(5 * BLH * 4);
  bf16* ob   = (bf16*)alloc(BLD * 2);
  bf16* ffb  = (bf16*)alloc(BLF * 2);
  bf16* wb_lnqkv = (bf16*)alloc((size_t)4194304 * 2);
  bf16* wb_fc1   = (bf16*)alloc((size_t)4194304 * 2);
  bf16* wb_fc2   = (bf16*)alloc((size_t)4194304 * 2);
  bf16* wb_saqkv = (bf16*)alloc((size_t)2097152 * 2);
  bf16* wb_safc1 = (bf16*)alloc((size_t)2097152 * 2);
  bf16* wb_safc2 = (bf16*)alloc((size_t)2097152 * 2);
  bf16* wb_caqkv = (bf16*)alloc((size_t)2097152 * 2);
  float* comb  = (float*)alloc((size_t)2 * 128 * 512 * 4);
  bf16* hcomb = (bf16*)alloc((size_t)2 * 128 * 512 * 2);
  bf16* sq    = (bf16*)alloc((size_t)2 * 128 * 512 * 2);
  bf16* sk    = (bf16*)alloc((size_t)2 * 128 * 512 * 2);
  bf16* sv    = (bf16*)alloc((size_t)2 * 128 * 512 * 2);
  bf16* sob   = (bf16*)alloc((size_t)2 * 128 * 512 * 2);
  bf16* sff   = (bf16*)alloc((size_t)2 * 128 * 2048 * 2);
  float* qfin = (float*)alloc((size_t)2 * 32 * 512 * 4);
  bf16* qfb   = (bf16*)alloc((size_t)2 * 32 * 512 * 2);
  bf16* cqb   = (bf16*)alloc((size_t)2 * 32 * 512 * 2);
  bf16* cob   = (bf16*)alloc((size_t)2 * 32 * 512 * 2);

  // weight bf16 conversion
  f2b_k<<<2048, 256, 0, stream>>>(w_lnqkv, wb_lnqkv, (size_t)4194304);
  f2b_k<<<2048, 256, 0, stream>>>(w_fc1,   wb_fc1,   (size_t)4194304);
  f2b_k<<<2048, 256, 0, stream>>>(w_fc2,   wb_fc2,   (size_t)4194304);
  f2b_k<<<2048, 256, 0, stream>>>(w_saqkv, wb_saqkv, (size_t)2097152);
  f2b_k<<<2048, 256, 0, stream>>>(w_safc1, wb_safc1, (size_t)2097152);
  f2b_k<<<2048, 256, 0, stream>>>(w_safc2, wb_safc2, (size_t)2097152);
  f2b_k<<<2048, 256, 0, stream>>>(w_caqkv, wb_caqkv, (size_t)2097152);

  posadd_k<<<16384, 128, 0, stream>>>(in_ctx, in_crd, in_pe, xbuf);
  copyf_k<<<256, 256, 0, stream>>>(in_q, qfin, (size_t)2 * 32 * 512);

  auto gemm = [&](const bf16* A, const bf16* Wp, const float* bias, const float* gam,
                  const float* res, float* oF, bf16* oB, int M, int N, int K, int gelu) {
    dim3 grid((N + 127) / 128, (M + 127) / 128);
    gemm_wmma_k<<<grid, 256, 0, stream>>>(A, Wp, bias, gam, res, oF, oB, M, N, K, gelu);
  };
  auto flash = [&](const bf16* Qp, const bf16* Kp, const bf16* Vp, bf16* Op, float* Ls,
                   int Tq, int Tkv, int qlen, int kvlen, int wseg, int rdil, int dil) {
    const int segH = dil ? (8192 / wseg) * 8 : 8;
    dim3 grid((qlen + 63) / 64, segH, 2);
    flash_attn_k<<<grid, 128, 0, stream>>>(Qp, Kp, Vp, Op, Ls, Tq, Tkv, qlen, kvlen,
                                           wseg, rdil, dil, 0.125f);
  };

  const int wsegs[5] = {512, 1024, 2048, 4096, 8192};
  const int rdils[5] = {1, 2, 4, 8, 16};

  for (int i = 0; i < 2; ++i) {
    for (int j = 0; j < 2; ++j) {
      const size_t lid = (size_t)i * 2 + j;
      layernorm_k<<<16384, 256, 0, stream>>>(xbuf,
          lnln + ((lid * 2 + 0) * 2 + 0) * 512, lnln + ((lid * 2 + 0) * 2 + 1) * 512,
          nullptr, hb);
      const size_t qw = lid * 4 * 262144;
      gemm(hb, wb_lnqkv + qw + 0 * 262144, b_lnqkv + (lid * 4 + 0) * 512, nullptr, nullptr, nullptr, qb, 16384, 512, 512, 0);
      gemm(hb, wb_lnqkv + qw + 1 * 262144, b_lnqkv + (lid * 4 + 1) * 512, nullptr, nullptr, nullptr, kb, 16384, 512, 512, 0);
      gemm(hb, wb_lnqkv + qw + 2 * 262144, b_lnqkv + (lid * 4 + 2) * 512, nullptr, nullptr, nullptr, vb, 16384, 512, 512, 0);
      fillf_k<<<1024, 256, 0, stream>>>(brl, -1.0e9f, (size_t)5 * BLH);
      for (int br = 0; br < 5; ++br)
        flash(qb, kb, vb, bro + (size_t)br * BLD, brl + (size_t)br * BLH,
              8192, 8192, 512, 512, wsegs[br], rdils[br], 1);
      combine_k<<<32768, 256, 0, stream>>>(bro, brl, ob);
      gemm(ob, wb_lnqkv + qw + 3 * 262144, b_lnqkv + (lid * 4 + 3) * 512, nullptr, xbuf, xbuf, nullptr, 16384, 512, 512, 0);
      layernorm_k<<<16384, 256, 0, stream>>>(xbuf,
          lnln + ((lid * 2 + 1) * 2 + 0) * 512, lnln + ((lid * 2 + 1) * 2 + 1) * 512,
          nullptr, hb);
      gemm(hb, wb_fc1 + lid * 1048576, b_fc1 + lid * 2048, nullptr, nullptr, nullptr, ffb, 16384, 2048, 512, 1);
      gemm(ffb, wb_fc2 + lid * 1048576, b_fc2 + lid * 512, nullptr, xbuf, xbuf, nullptr, 16384, 512, 2048, 0);
    }
    // final context LN -> fp32 state + bf16 ctx for cross-attn K/V
    layernorm_k<<<16384, 256, 0, stream>>>(xbuf,
        lnfinal + ((size_t)i * 2 + 0) * 512, lnfinal + ((size_t)i * 2 + 1) * 512, xbuf, hb);

    // self-attention over [querys ; instructs]
    combbuild_k<<<256, 128, 0, stream>>>(qfin, in_inst, comb);
    layernorm_k<<<256, 256, 0, stream>>>(comb,
        saln + (((size_t)i * 2 + 0) * 2 + 0) * 512, saln + (((size_t)i * 2 + 0) * 2 + 1) * 512,
        nullptr, hcomb);
    const size_t sw = (size_t)i * 4 * 262144;
    gemm(hcomb, wb_saqkv + sw + 0 * 262144, b_saqkv + ((size_t)i * 4 + 0) * 512, nullptr, nullptr, nullptr, sq, 256, 512, 512, 0);
    gemm(hcomb, wb_saqkv + sw + 1 * 262144, b_saqkv + ((size_t)i * 4 + 1) * 512, nullptr, nullptr, nullptr, sk, 256, 512, 512, 0);
    gemm(hcomb, wb_saqkv + sw + 2 * 262144, b_saqkv + ((size_t)i * 4 + 2) * 512, nullptr, nullptr, nullptr, sv, 256, 512, 512, 0);
    flash(sq, sk, sv, sob, nullptr, 128, 128, 128, 128, 0, 1, 0);
    gemm(sob, wb_saqkv + sw + 3 * 262144, b_saqkv + ((size_t)i * 4 + 3) * 512,
         sagamma + ((size_t)i * 2 + 0) * 512, comb, comb, nullptr, 256, 512, 512, 0);
    layernorm_k<<<256, 256, 0, stream>>>(comb,
        saln + (((size_t)i * 2 + 1) * 2 + 0) * 512, saln + (((size_t)i * 2 + 1) * 2 + 1) * 512,
        nullptr, hcomb);
    gemm(hcomb, wb_safc1 + (size_t)i * 1048576, b_safc1 + (size_t)i * 2048, nullptr, nullptr, nullptr, sff, 256, 2048, 512, 1);
    gemm(sff, wb_safc2 + (size_t)i * 1048576, b_safc2 + (size_t)i * 512,
         sagamma + ((size_t)i * 2 + 1) * 512, comb, comb, nullptr, 256, 512, 2048, 0);

    // cross-attention: querys attend to contexts
    extractq_k<<<64, 128, 0, stream>>>(comb, qfin, qfb);
    const size_t cw = (size_t)i * 4 * 262144;
    gemm(qfb, wb_caqkv + cw + 0 * 262144, b_caqkv + ((size_t)i * 4 + 0) * 512, nullptr, nullptr, nullptr, cqb, 64, 512, 512, 0);
    gemm(hb,  wb_caqkv + cw + 1 * 262144, b_caqkv + ((size_t)i * 4 + 1) * 512, nullptr, nullptr, nullptr, kb, 16384, 512, 512, 0);
    gemm(hb,  wb_caqkv + cw + 2 * 262144, b_caqkv + ((size_t)i * 4 + 2) * 512, nullptr, nullptr, nullptr, vb, 16384, 512, 512, 0);
    flash(cqb, kb, vb, cob, nullptr, 32, 8192, 32, 8192, 0, 1, 0);
    gemm(cob, wb_caqkv + cw + 3 * 262144, b_caqkv + ((size_t)i * 4 + 3) * 512, nullptr, qfin, qfin, nullptr, 64, 512, 512, 0);
    layernorm_k<<<64, 256, 0, stream>>>(qfin,
        caln + ((size_t)i * 2 + 0) * 512, caln + ((size_t)i * 2 + 1) * 512,
        qfin, (i == 1) ? (bf16*)d_out : nullptr);
  }
}
